// self_attn_init_40200893890894
// MI455X (gfx1250) — matrix operand order repaired and checked by reading, not compiled, not run on hardware
//
#include <hip/hip_runtime.h>
#include <hip/hip_bf16.h>

typedef __attribute__((ext_vector_type(16))) _Float16 v16h;
typedef __attribute__((ext_vector_type(8)))  _Float16 v8h;
typedef __attribute__((ext_vector_type(8)))  float    v8f;
typedef unsigned int u32x4 __attribute__((ext_vector_type(4)));
typedef int          i32x8 __attribute__((ext_vector_type(8)));
typedef int          i32x4 __attribute__((ext_vector_type(4)));

#define N_TOT 8192
#define DH    512
#define MBLK  64
#define KVB   64

#if __has_builtin(__builtin_amdgcn_tensor_load_to_lds)
#define HAVE_TDM 1
#else
#define HAVE_TDM 0
#endif

__device__ __forceinline__ v16h cat8(v8h lo, v8h hi) {
  return __builtin_shufflevector(lo, hi, 0,1,2,3,4,5,6,7,8,9,10,11,12,13,14,15);
}

// A fragment (16x32 f16, MxK): lane m=lane&15, hi=lane>>4;
// comps 0-7 -> K=k0+hi*8+j, comps 8-15 -> K=k0+16+hi*8+j (ISA 7.12.2).
__device__ __forceinline__ v16h load_a_frag(const _Float16* buf, int stride,
                                            int row0, int k0, int lane) {
  int m  = lane & 15;
  int hi = lane >> 4;
  const _Float16* p = buf + (size_t)(row0 + m) * stride + k0 + hi * 8;
  v8h lo  = *(const v8h*)(p);
  v8h hi8 = *(const v8h*)(p + 16);
  return cat8(lo, hi8);
}

// B fragment (32x16 f16, KxN), src stores one row per output column:
// comps 0-7 -> src[n0+n][k0+hi*8+j], comps 8-15 -> src[n0+n][k0+16+hi*8+j].
__device__ __forceinline__ v16h load_b_frag(const _Float16* buf, int stride,
                                            int n0, int k0, int lane) {
  int n  = lane & 15;
  int hi = lane >> 4;
  const _Float16* p = buf + (size_t)(n0 + n) * stride + k0 + hi * 8;
  v8h lo  = *(const v8h*)(p);
  v8h hi8 = *(const v8h*)(p + 16);
  return cat8(lo, hi8);
}

// Low 32 bits of a generic LDS address are the LDS byte offset (aperture rule).
__device__ __forceinline__ unsigned lds_off(const void* p) {
  return (unsigned)(uintptr_t)p;
}

// gfx1250 async copy global -> LDS, 16B per lane, tracked by ASYNCcnt.
__device__ __forceinline__ void async_load_b128(unsigned lds, const void* g) {
  asm volatile("global_load_async_to_lds_b128 %0, %1, off"
               :: "v"(lds), "v"(g) : "memory");
}

__device__ __forceinline__ void wait_async0() {
#if __has_builtin(__builtin_amdgcn_s_wait_asynccnt)
  __builtin_amdgcn_s_wait_asynccnt(0);
#else
  asm volatile("s_wait_asynccnt 0x0" ::: "memory");
#endif
}
__device__ __forceinline__ void wait_async8() {
#if __has_builtin(__builtin_amdgcn_s_wait_asynccnt)
  __builtin_amdgcn_s_wait_asynccnt(8);
#else
  asm volatile("s_wait_asynccnt 0x8" ::: "memory");
#endif
}

#if HAVE_TDM
__device__ __forceinline__ void wait_tensor0() {
#if __has_builtin(__builtin_amdgcn_s_wait_tensorcnt)
  __builtin_amdgcn_s_wait_tensorcnt(0);
#else
  asm volatile("s_wait_tensorcnt 0x0" ::: "memory");
#endif
}
__device__ __forceinline__ void wait_tensor1() {
#if __has_builtin(__builtin_amdgcn_s_wait_tensorcnt)
  __builtin_amdgcn_s_wait_tensorcnt(1);
#else
  asm volatile("s_wait_tensorcnt 0x1" ::: "memory");
#endif
}

// TDM 2D tile load: tile (tile_d0 x tile_d1) halfs from a tensor with row
// length tensor_d0 / row stride stride0 (halfs), into LDS with per-row
// padding (pad codes per D# group1: interval 2^(c+1) DW, amount c+1 DW).
__device__ __forceinline__ void tdm_load_2d(unsigned lds, const void* gaddr,
                                            unsigned tensor_d0, unsigned tensor_d1,
                                            unsigned tile_d0, unsigned tile_d1,
                                            unsigned long long stride0,
                                            unsigned pad_int_code, unsigned pad_amt_code) {
  unsigned long long ga = (unsigned long long)(uintptr_t)gaddr;
  unsigned long long s0 = stride0, s1 = stride0;
  u32x4 g0;
  g0[0] = 1u;                                    // count=1 (valid user descriptor)
  g0[1] = lds;                                   // lds_addr
  g0[2] = (unsigned)ga;                          // global_addr[31:0]
  g0[3] = (unsigned)((ga >> 32) & 0x1ffffffu) | (2u << 30);  // addr[56:32] | type=2
  i32x8 g1;
  g1[0] = (int)((1u << 16) | (1u << 20) |        // data_size=2B, pad_enable
                (pad_int_code << 22) | (pad_amt_code << 25));
  g1[1] = (int)((tensor_d0 & 0xffffu) << 16);
  g1[2] = (int)(((tensor_d0 >> 16) & 0xffffu) | ((tensor_d1 & 0xffffu) << 16));
  g1[3] = (int)(((tensor_d1 >> 16) & 0xffffu) | ((tile_d0 & 0xffffu) << 16));
  g1[4] = (int)(tile_d1 & 0xffffu);
  g1[5] = (int)(unsigned)s0;
  g1[6] = (int)(((s0 >> 32) & 0xffffu) | ((s1 & 0xffffu) << 16));
  g1[7] = (int)((s1 >> 16) & 0xffffffffu);
  i32x4 z4 = {0, 0, 0, 0};
  i32x8 z8 = {0, 0, 0, 0, 0, 0, 0, 0};
  __builtin_amdgcn_tensor_load_to_lds(g0, g1, z4, z4, z8, 0);
}
#endif  // HAVE_TDM

// ---------------- Kernel 1: keys/queries projection (D_IN=6) ----------------
__global__ void qk_proj_kernel(const float* __restrict__ nodes,
                               const float* __restrict__ centre,
                               const float* __restrict__ Wq, const float* __restrict__ bq,
                               const float* __restrict__ Wk, const float* __restrict__ bk,
                               float* __restrict__ keys_f32,
                               _Float16* __restrict__ k16,
                               _Float16* __restrict__ q16) {
  int gid = blockIdx.x * blockDim.x + threadIdx.x;
  if (gid >= N_TOT * DH) return;
  int n = gid >> 9;
  int d = gid & (DH - 1);
  float x0 = nodes[n * 3 + 0], x1 = nodes[n * 3 + 1], x2 = nodes[n * 3 + 2];
  float c0 = centre[0], c1 = centre[1], c2 = centre[2];
  const float* wq = Wq + d * 6;
  const float* wk = Wk + d * 6;
  float q = bq[d] + x0 * wq[0] + x1 * wq[1] + x2 * wq[2]
                  + c0 * wq[3] + c1 * wq[4] + c2 * wq[5];
  float k = bk[d] + x0 * wk[0] + x1 * wk[1] + x2 * wk[2]
                  + c0 * wk[3] + c1 * wk[4] + c2 * wk[5];
  keys_f32[gid] = k;
  k16[gid] = (_Float16)k;
  q16[gid] = (_Float16)q;
}

// ---------------- Kernel 2: Wv f32 -> f16 ----------------
__global__ void wv_cvt_kernel(const float* __restrict__ Wv, _Float16* __restrict__ Wv16) {
  int gid = blockIdx.x * blockDim.x + threadIdx.x;
  if (gid < DH * 2 * DH) Wv16[gid] = (_Float16)Wv[gid];
}

// ------------- Kernel 3: WMMA GEMM  C[n][d] = sum_k A[n][k] * W[d][colbase+k] -------------
// mode 0: vT16[d][n] = (f16)acc        (Vq^T, transposed for flash V staging)
// mode 1: out32 = keys + acc + bv      (base = keys + keys @ Wv_k^T + bv)
__global__ __launch_bounds__(256)
void gemm_wmma_kernel(const _Float16* __restrict__ A16,
                      const _Float16* __restrict__ W16,
                      int colbase, int mode,
                      const float* __restrict__ keys_f32,
                      const float* __restrict__ bv,
                      _Float16* __restrict__ out16T,
                      float* __restrict__ out32) {
  int wave = threadIdx.x >> 5;
  int lane = threadIdx.x & 31;
  int tile = blockIdx.x * 8 + wave;        // 16384 tiles total
  int tm = tile >> 5;                      // 512 row tiles
  int tn = tile & 31;                      // 32 col tiles

  v8f acc;
  #pragma unroll
  for (int e = 0; e < 8; ++e) acc[e] = 0.0f;

  #pragma unroll
  for (int kc = 0; kc < DH; kc += 32) {
    v16h a = load_a_frag(A16, DH, tm * 16, kc, lane);
    v16h b = load_b_frag(W16, 2 * DH, tn * 16, colbase + kc, lane);
    acc = __builtin_amdgcn_wmma_f32_16x16x32_f16(false, a, false, b,
                                                 (short)0, acc, false, false);
  }

  int hi = lane >> 4, cn = lane & 15;
  int col = tn * 16 + cn;
  if (mode == 0) {
    #pragma unroll
    for (int r = 0; r < 8; ++r)
      out16T[(size_t)col * N_TOT + (tm * 16 + 8 * hi + r)] = (_Float16)acc[r];
  } else {
    float b = bv[col];
    #pragma unroll
    for (int r = 0; r < 8; ++r) {
      size_t off = (size_t)(tm * 16 + 8 * hi + r) * DH + col;
      out32[off] = keys_f32[off] + acc[r] + b;
    }
  }
}

// ---------------- Kernel 4: flash attention ----------------
// out = base + softmax(K Q^T) @ V, K=keys, Q=queries f16 [8192][512],
// V supplied transposed: vT[d][n] f16 [512][8192].
// Workgroup: 64 key rows, 16 waves. KV blocks of 64.
// Staging: Q via global_load_async_to_lds_b128 (ASYNCcnt), K/V via TDM
// (TENSORcnt) when available; V DMA overlaps the S-phase WMMAs.
__global__ __launch_bounds__(512, 1)
void flash_attn_kernel(const _Float16* __restrict__ Kf,
                       const _Float16* __restrict__ Qf,
                       const _Float16* __restrict__ VfT,
                       const float* __restrict__ basep,
                       float* __restrict__ outp) {
  __shared__ _Float16 Kt[MBLK][520];     // keys tile (row pad: 256 DW + 4 DW)
  __shared__ _Float16 Qt[KVB][520];      // queries tile
  __shared__ _Float16 VtT[DH][72];       // V^T tile (row pad: 32 DW + 4 DW)
  __shared__ _Float16 Pt[MBLK][72];      // softmax probs f16
  __shared__ float    Ssh[MBLK][66];     // raw scores
  __shared__ float    row_max[MBLK], row_sum[MBLK], row_scale[MBLK];

  const int tid  = threadIdx.x;
  const int lane = tid & 31;
  const int wv   = tid >> 5;      // wave 0..15
  const int rt   = wv >> 2;       // row tile 0..3
  const int cq   = wv & 3;        // S col tile / O d-group 0..3
  const int hi   = lane >> 4;
  const int cn   = lane & 15;
  const int m0   = blockIdx.x * MBLK;

  // ---- stage K tile (once) ----
#if HAVE_TDM
  if (wv == 0)
    tdm_load_2d(lds_off(&Kt[0][0]), Kf + (size_t)m0 * DH,
                DH, N_TOT, DH, MBLK, DH, /*pad 256DW*/7, /*4DW*/3);
#else
  #pragma unroll
  for (int it = 0; it < 8; ++it) {
    int idx = tid + it * 512;
    int row = idx >> 6, c8 = idx & 63;
    async_load_b128(lds_off(&Kt[row][c8 * 8]),
                    Kf + (size_t)(m0 + row) * DH + c8 * 8);
  }
#endif
  if (tid < MBLK) { row_max[tid] = -3.0e38f; row_sum[tid] = 0.0f; }

  v8f acc[8];
  #pragma unroll
  for (int t = 0; t < 8; ++t)
    #pragma unroll
    for (int e = 0; e < 8; ++e) acc[t][e] = 0.0f;

  for (int kv = 0; kv < N_TOT; kv += KVB) {
    if (kv + KVB < N_TOT)
      __builtin_prefetch(Qf + (size_t)(kv + KVB) * DH, 0, 1);

    // ---- issue Q (async, all waves) and V (TDM wave 0 / async fallback) ----
    #pragma unroll
    for (int it = 0; it < 8; ++it) {
      int idx = tid + it * 512;
      int row = idx >> 6, c8 = idx & 63;
      async_load_b128(lds_off(&Qt[row][c8 * 8]),
                      Qf + (size_t)(kv + row) * DH + c8 * 8);
    }
#if HAVE_TDM
    if (wv == 0)
      tdm_load_2d(lds_off(&VtT[0][0]), VfT + kv,
                  N_TOT, DH, KVB, DH, N_TOT, /*pad 32DW*/4, /*4DW*/3);
    wait_async0();               // Q in LDS
    if (wv == 0) wait_tensor1(); // first iter: K done, V still in flight
#else
    #pragma unroll
    for (int it = 0; it < 8; ++it) {
      int idx = tid + it * 512;
      int row = idx >> 3, c8 = idx & 7;
      async_load_b128(lds_off(&VtT[row][c8 * 8]),
                      VfT + (size_t)row * N_TOT + kv + c8 * 8);
    }
    wait_async8();               // in-order: K(first iter)+Q done, V in flight
#endif
    __syncthreads();

    // ---- S tile: S[rt][cq] = K_rows(rt) . Q_rows(cq), Kdim=512 (V DMA overlaps) ----
    v8f s;
    #pragma unroll
    for (int e = 0; e < 8; ++e) s[e] = 0.0f;
    #pragma unroll
    for (int kc = 0; kc < DH; kc += 32) {
      v16h a = load_a_frag(&Kt[0][0], 520, rt * 16, kc, lane);
      v16h b = load_b_frag(&Qt[0][0], 520, cq * 16, kc, lane);
      s = __builtin_amdgcn_wmma_f32_16x16x32_f16(false, a, false, b,
                                                 (short)0, s, false, false);
    }
    #pragma unroll
    for (int r = 0; r < 8; ++r)
      Ssh[rt * 16 + 8 * hi + r][cq * 16 + cn] = s[r];
    __syncthreads();

    // ---- online softmax stats: thread t < 64 owns row t ----
    if (tid < MBLK) {
      float mold = row_max[tid];
      float mx = mold;
      #pragma unroll 8
      for (int j = 0; j < KVB; ++j) mx = fmaxf(mx, Ssh[tid][j]);
      float scale = __expf(mold - mx);
      float sum = row_sum[tid] * scale;
      #pragma unroll 8
      for (int j = 0; j < KVB; ++j) {
        float p = __expf(Ssh[tid][j] - mx);
        Pt[tid][j] = (_Float16)p;
        sum += p;
      }
      row_max[tid] = mx; row_sum[tid] = sum; row_scale[tid] = scale;
    }
#if HAVE_TDM
    if (wv == 0) wait_tensor0();  // V tile landed
#else
    wait_async0();
#endif
    __syncthreads();

    // ---- rescale O, then O += P @ V for this wave's 8 d-tiles ----
    float scl[8];
    #pragma unroll
    for (int r = 0; r < 8; ++r) scl[r] = row_scale[rt * 16 + 8 * hi + r];
    #pragma unroll
    for (int t = 0; t < 8; ++t)
      #pragma unroll
      for (int r = 0; r < 8; ++r) acc[t][r] *= scl[r];

    #pragma unroll
    for (int kc = 0; kc < KVB; kc += 32) {
      v16h a = load_a_frag(&Pt[0][0], 72, rt * 16, kc, lane);
      #pragma unroll
      for (int t = 0; t < 8; ++t) {
        int d0 = (cq * 8 + t) * 16;
        v16h b = load_b_frag(&VtT[0][0], 72, d0, kc, lane);
        acc[t] = __builtin_amdgcn_wmma_f32_16x16x32_f16(false, a, false, b,
                                                        (short)0, acc[t], false, false);
      }
    }
    __syncthreads();
  }

  // ---- epilogue: out = base + acc / row_sum ----
  float inv[8];
  #pragma unroll
  for (int r = 0; r < 8; ++r) inv[r] = 1.0f / row_sum[rt * 16 + 8 * hi + r];
  #pragma unroll
  for (int t = 0; t < 8; ++t) {
    int d = (cq * 8 + t) * 16 + cn;
    #pragma unroll
    for (int r = 0; r < 8; ++r) {
      size_t off = (size_t)(m0 + rt * 16 + 8 * hi + r) * DH + d;
      outp[off] = basep[off] + acc[t][r] * inv[r];
    }
  }
}

extern "C" void kernel_launch(void* const* d_in, const int* in_sizes, int n_in,
                              void* d_out, int out_size, void* d_ws, size_t ws_size,
                              hipStream_t stream) {
  const float* nodes  = (const float*)d_in[0];
  const float* centre = (const float*)d_in[1];
  const float* Wq     = (const float*)d_in[2];
  const float* bq     = (const float*)d_in[3];
  const float* Wk     = (const float*)d_in[4];
  const float* bk     = (const float*)d_in[5];
  const float* Wv     = (const float*)d_in[6];
  const float* bv     = (const float*)d_in[7];
  float* out = (float*)d_out;

  char* ws = (char*)d_ws;
  const size_t MB = 1024 * 1024;
  float*    keys_f32 = (float*)(ws);              // 16 MB
  float*    base_f32 = (float*)(ws + 16 * MB);    // 16 MB
  _Float16* k16      = (_Float16*)(ws + 32 * MB); // 8 MB
  _Float16* q16      = (_Float16*)(ws + 40 * MB); // 8 MB
  _Float16* vT16     = (_Float16*)(ws + 48 * MB); // 8 MB (Vq transposed [512][8192])
  _Float16* Wv16     = (_Float16*)(ws + 56 * MB); // 1 MB

  qk_proj_kernel<<<(N_TOT * DH) / 256, 256, 0, stream>>>(
      nodes, centre, Wq, bq, Wk, bk, keys_f32, k16, q16);
  wv_cvt_kernel<<<(DH * 2 * DH) / 256, 256, 0, stream>>>(Wv, Wv16);
  // Vq^T = (queries @ Wv_q^T)^T  (f16, transposed)
  gemm_wmma_kernel<<<2048, 256, 0, stream>>>(q16, Wv16, 0, 0,
                                             nullptr, nullptr, vT16, nullptr);
  // base = keys + keys @ Wv_k^T + bv  (f32)
  gemm_wmma_kernel<<<2048, 256, 0, stream>>>(k16, Wv16, DH, 1,
                                             keys_f32, bv, nullptr, base_f32);
  flash_attn_kernel<<<N_TOT / MBLK, 512, 0, stream>>>(k16, q16, vT16, base_f32, out);
}
